// CompositionNet_4217657885290
// MI455X (gfx1250) — matrix operand order repaired and checked by reading, not compiled, not run on hardware
//
#include <hip/hip_runtime.h>
#include <hip/hip_bf16.h>

// =====================================================================
// CompositionNet forward for MI455X (gfx1250, wave32, WMMA bf16 + TDM)
//
// Dense GEMMs -> v_wmma_f32_16x16x32_bf16 (bf16 operands, f32 accum).
// LDS tiles staged by the Tensor Data Mover (tensor_load_to_lds, one
// issuing wave per block, s_wait_tensorcnt + workgroup barrier), with a
// manual global->LDS fallback when the TDM builtin is unavailable.
//
// d_in layout (setup_inputs() dict order, params nested in file order):
//  0 atom_weights (N) f32, 1 orig_atom_fea (N*200) f32,
//  2 self_fea_idx (M) i32, 3 nbr_fea_idx (M) i32, 4 crystal_atom_idx (N) i32
//  5 embed W (200x128), 6 embed b (128)
//  7 + g*10 (g=0..2): lin_in W(256x512), b(512), lin_out W(512x128), b(128),
//                     gate W0(128x384), b0, gate W1(384x128), b1,
//                     gate W2(128x1), b2
//  37..44: cry_gate W0(128x640) b0 W1(640x384) b1 W2(384x128) b2 W3(128x1) b3
//  45..54: out W0(128x896) b0 W1(896x640) b1 W2(640x384) b2 W3(384x128) b3
//          W4(128x2) b4
// d_out: (C x 2) f32
// =====================================================================

typedef __attribute__((ext_vector_type(16))) __bf16 v16bf;
typedef __attribute__((ext_vector_type(8)))  float  v8f;

#define GN 65536
#define GM 262144
#define GC 8192

#if defined(__has_builtin)
#if __has_builtin(__builtin_amdgcn_tensor_load_to_lds) && \
    __has_builtin(__builtin_amdgcn_s_wait_tensorcnt)
#define USE_TDM 1
#endif
#endif
#ifndef USE_TDM
#define USE_TDM 0
#endif

__device__ __forceinline__ unsigned short f2bf(float f) {
  unsigned int u = __float_as_uint(f);
  u += 0x7FFFu + ((u >> 16) & 1u);  // round-to-nearest-even
  return (unsigned short)(u >> 16);
}
__device__ __forceinline__ float bf2f(unsigned short h) {
  return __uint_as_float(((unsigned int)h) << 16);
}
__device__ __forceinline__ void atomicMaxFloat(float* addr, float val) {
  // init value must be -inf (0xFF800000); valid for any pos/neg mix.
  if (val >= 0.f) atomicMax((int*)addr, __float_as_int(val));
  else            atomicMin((unsigned int*)addr, __float_as_uint(val));
}

// ---------------------------------------------------------------------
// Tensor Data Mover: async 2D tile load global -> LDS.
// D# per CDNA5 ISA ch.8: group0 = {ctl, lds_addr, gaddr, type=2},
// group1 = {mask/size/pad, tensor_dim0/1, tile_dim0/1, dim0_stride}.
// pad_enable with pad_interval=3 (16 DWORDs) / pad_amount=3 (4 DWORDs)
// turns a 32-elem bf16 row (64B) into a 40-elem (80B) LDS pitch = our
// bank-padded tile stride. tensor_dim0 = remaining K gives zero-fill on
// the K tail. Tracked by TENSORcnt.
// This toolchain exposes the 6-arg builtin:
//   (uint32x4 g0, int32x8 g1, int32x4 g2, int32x4 g3, int32x8, i32 cpol)
// Groups 2/3 (and the extra group) are zero for 2D tensors.
// ---------------------------------------------------------------------
#if USE_TDM
typedef __attribute__((ext_vector_type(4))) unsigned int u32x4;
typedef __attribute__((ext_vector_type(8))) int i32x8;
typedef __attribute__((ext_vector_type(4))) int i32x4;

__device__ __forceinline__ void tdm_load_tile(
    unsigned lds_off,                 // LDS byte address (low 32 of generic)
    const unsigned short* gptr,       // global tile start
    unsigned rem_k,                   // valid elems along dim0 (K - k0)
    unsigned rows_total,              // tensor_dim1 (all rows valid)
    unsigned tile_rows,               // tile dim1
    unsigned long long pitch_elems) { // tensor_dim0_stride (= K)
  const unsigned long long ga = (unsigned long long)gptr;
  u32x4 g0 = {1u,                                    // count=1, user mode
              lds_off,                               // lds_addr
              (unsigned)(ga & 0xFFFFFFFFu),          // global_addr[31:0]
              (unsigned)((ga >> 32) & 0x01FFFFFFu) | (2u << 30)};  // type=2
  const unsigned dw0 = (1u << 16)      // data_size = 2 bytes
                     | (1u << 20)      // pad_enable
                     | (3u << 22)      // pad_interval: 16 DWORDs
                     | (3u << 25);     // pad_amount:   4 DWORDs
  i32x8 g1 = {(int)dw0,
              (int)((rem_k & 0xFFFFu) << 16),                       // dim0 lo
              (int)((rem_k >> 16) | ((rows_total & 0xFFFFu) << 16)),// dim0 hi|dim1 lo
              (int)((rows_total >> 16) | (32u << 16)),              // dim1 hi|tile_dim0
              (int)(tile_rows & 0xFFFFu),                           // tile_dim1 (tile_dim2=0)
              (int)(pitch_elems & 0xFFFFFFFFu),                     // dim0_stride lo
              (int)((pitch_elems >> 32) & 0xFFFFu),                 // dim0_stride hi
              0};
  const i32x4 z4 = {0, 0, 0, 0};
  const i32x8 z8 = {0, 0, 0, 0, 0, 0, 0, 0};
  __builtin_amdgcn_tensor_load_to_lds(g0, g1, z4, z4, z8, 0);
}
#endif

// ---------------------------------------------------------------------
// Generic WMMA GEMM:  out = act(A(MxK bf16) @ WT(NxK bf16)^T + bias)
// Block tile 128(M) x 64(N), 8 waves, each wave 32x32 (2x2 wmma tiles),
// K stepped by 32. Mrows%128==0, N%64==0 required.
// ---------------------------------------------------------------------
#define TM 128
#define TN 64
#define AS 40  // LDS row pitch (elems): 32 + 8 pad (80B = 20 DWORDs)
#define BS 40

template <bool RELU, bool O16, bool O32, bool KTAIL>
__global__ __launch_bounds__(256) void gemm_bf16_kernel(
    const unsigned short* __restrict__ A,    // Mrows x K, row-major bf16
    const unsigned short* __restrict__ WT,   // N x K, row-major bf16
    const float* __restrict__ bias,          // N
    unsigned short* __restrict__ out16,      // Mrows x N bf16
    float* __restrict__ out32,               // Mrows x N f32
    int Mrows, int N, int K) {
  __shared__ unsigned short lA[TM * AS];
  __shared__ unsigned short lB[TN * BS];
  const int tid  = threadIdx.x;
  const int wave = tid >> 5;
  const int lane = tid & 31;
  const int wr   = wave & 3;   // 4 wave-rows * 32 = 128
  const int wc   = wave >> 2;  // 2 wave-cols * 32 = 64
  const long blockM = (long)blockIdx.x * TM;
  const long blockN = (long)blockIdx.y * TN;

#if USE_TDM
  const unsigned ldsA = (unsigned)(size_t)(void*)lA;
  const unsigned ldsB = (unsigned)(size_t)(void*)lB;
#endif

  v8f acc[2][2] = {};

  for (int k0 = 0; k0 < K; k0 += 32) {
#if USE_TDM
    // ---- async TDM staging: one issuing wave, DMA writes LDS ----
    if (wave == 0) {
      tdm_load_tile(ldsA, A + blockM * (long)K + k0, (unsigned)(K - k0),
                    (unsigned)Mrows, TM, (unsigned long long)K);
      tdm_load_tile(ldsB, WT + blockN * (long)K + k0, (unsigned)(K - k0),
                    (unsigned)N, TN, (unsigned long long)K);
      __builtin_amdgcn_s_wait_tensorcnt(0);
    }
#else
    // ---- manual staging: A 128x32 (16 bf16/thread), B 64x32 (8/thread) ----
    {
      const int row = tid >> 1, seg = tid & 1;
      const unsigned short* src = A + (blockM + row) * (long)K + k0 + seg * 16;
      unsigned short* dst = &lA[row * AS + seg * 16];
      if (!KTAIL || k0 + 32 <= K) {
        *(uint4*)(dst)     = *(const uint4*)(src);
        *(uint4*)(dst + 8) = *(const uint4*)(src + 8);
      } else {
#pragma unroll
        for (int j = 0; j < 16; ++j) {
          const int k = k0 + seg * 16 + j;
          dst[j] = (k < K) ? src[j] : (unsigned short)0;
        }
      }
      if (k0 + 32 < K) __builtin_prefetch(src + 32, 0, 1);
    }
    {
      const int n = tid >> 2, seg = tid & 3;
      const unsigned short* src = WT + (blockN + n) * (long)K + k0 + seg * 8;
      unsigned short* dst = &lB[n * BS + seg * 8];
      if (!KTAIL || k0 + 32 <= K) {
        *(uint4*)(dst) = *(const uint4*)(src);
      } else {
#pragma unroll
        for (int j = 0; j < 8; ++j) {
          const int k = k0 + seg * 8 + j;
          dst[j] = (k < K) ? src[j] : (unsigned short)0;
        }
      }
    }
#endif
    __syncthreads();

    // ---- fragments per the CDNA5 bf16 A (16x32) / B (32x16) layouts ----
    union V16 { v16bf v; uint4 q[2]; };
    V16 afr[2], bfr[2];
    {
      // A: lane l -> row m=l%16; chunks [kb..kb+7],[kb+16..kb+23], kb=(l/16)*8
      const int mrow = lane & 15;
      const int kb   = (lane >> 4) * 8;
#pragma unroll
      for (int mi = 0; mi < 2; ++mi) {
        const unsigned short* p = &lA[(wr * 32 + mi * 16 + mrow) * AS];
        afr[mi].q[0] = *(const uint4*)(p + kb);
        afr[mi].q[1] = *(const uint4*)(p + kb + 16);
      }
      // B: lane l -> col n=l%16; chunks [kb..kb+7],[kb+16..kb+23], kb=(l/16)*8
      const int kbB = (lane >> 4) * 8;
#pragma unroll
      for (int ni = 0; ni < 2; ++ni) {
        const unsigned short* p = &lB[(wc * 32 + ni * 16 + (lane & 15)) * BS + kbB];
        bfr[ni].q[0] = *(const uint4*)(p);
        bfr[ni].q[1] = *(const uint4*)(p + 16);
      }
    }
#pragma unroll
    for (int mi = 0; mi < 2; ++mi)
#pragma unroll
      for (int ni = 0; ni < 2; ++ni)
        acc[mi][ni] = __builtin_amdgcn_wmma_f32_16x16x32_bf16(
            false, afr[mi].v, false, bfr[ni].v, (short)0, acc[mi][ni], false, false);
    __syncthreads();
  }

  // ---- epilogue: C/D layout: lane l, vgpr v -> m=(l/16)*8+v, n=l%16 ----
#pragma unroll
  for (int mi = 0; mi < 2; ++mi) {
#pragma unroll
    for (int ni = 0; ni < 2; ++ni) {
      const long n = blockN + wc * 32 + ni * 16 + (lane & 15);
      const float bv = bias[n];
#pragma unroll
      for (int v = 0; v < 8; ++v) {
        const long m = blockM + wr * 32 + mi * 16 + (lane >> 4) * 8 + v;
        float val = acc[mi][ni][v] + bv;
        if (RELU) val = fmaxf(val, 0.f);
        if constexpr (O32) out32[m * N + n] = val;
        if constexpr (O16) out16[m * N + n] = f2bf(val);
      }
    }
  }
}

// ---------------------------------------------------------------------
// Tiny N output layers (N=1 or 2): one wave per row, lane-strided dot.
// ---------------------------------------------------------------------
__global__ __launch_bounds__(256) void rowdot_kernel(
    const unsigned short* __restrict__ X, const float* __restrict__ W,
    const float* __restrict__ bias, float* __restrict__ out,
    int rows, int K, int nout) {
  const int wave = blockIdx.x * (blockDim.x >> 5) + (threadIdx.x >> 5);
  const int lane = threadIdx.x & 31;
  if (wave >= rows) return;
  const unsigned short* x = X + (long)wave * K;
  for (int o = 0; o < nout; ++o) {
    float s = 0.f;
    for (int k = lane; k < K; k += 32) s += bf2f(x[k]) * W[(long)k * nout + o];
#pragma unroll
    for (int off = 16; off; off >>= 1) s += __shfl_down(s, off, 32);
    if (lane == 0) out[(long)wave * nout + o] = s + bias[o];
  }
}

// ---- weight transpose + f32->bf16 convert: W (KxN) -> WT (NxK) ----
__global__ void wcvt_kernel(const float* __restrict__ W,
                            unsigned short* __restrict__ WT, int K, int N) {
  const long t = blockIdx.x * 256ll + threadIdx.x;
  if (t >= (long)K * N) return;
  const int n = (int)(t / K), k = (int)(t % K);
  WT[t] = f2bf(W[(long)k * N + n]);
}

__global__ void cvt_bf16_kernel(const float* __restrict__ in,
                                unsigned short* __restrict__ out, long n) {
  const long t = blockIdx.x * 256ll + threadIdx.x;
  if (t < n) out[t] = f2bf(in[t]);
}

__global__ void fill_kernel(float* __restrict__ p, float v, long n) {
  const long t = blockIdx.x * 256ll + threadIdx.x;
  if (t < n) p[t] = v;
}

// ---- edge gather-concat: e_in[m] = [afea16[self[m]], afea16[nbr[m]]] ----
__global__ void gather_kernel(const unsigned short* __restrict__ afea16,
                              const int* __restrict__ self_idx,
                              const int* __restrict__ nbr_idx,
                              unsigned short* __restrict__ e_in, int M) {
  const long t = blockIdx.x * 256ll + threadIdx.x;  // M*32 threads, 8 bf16 each
  if (t >= (long)M * 32) return;
  const int m = (int)(t >> 5), seg = (int)(t & 31);
  const int src = (seg < 16) ? self_idx[m] : nbr_idx[m];
  *(uint4*)(e_in + (long)m * 256 + seg * 8) =
      *(const uint4*)(afea16 + (long)src * 128 + (seg & 15) * 8);
}

// ---- segment-softmax attention (3 phases + pooled scatter) ----
__global__ void attn_max_kernel(const float* __restrict__ logit,
                                const int* __restrict__ idx,
                                float* __restrict__ segmax, int E) {
  const int e = blockIdx.x * 256 + threadIdx.x;
  if (e >= E) return;
  atomicMaxFloat(&segmax[idx[e]], logit[e]);
}
__global__ void attn_num_kernel(float* __restrict__ logit,
                                const int* __restrict__ idx,
                                const int* __restrict__ widx,
                                const float* __restrict__ weights,
                                const float* __restrict__ segmax,
                                float* __restrict__ segden, int E) {
  const int e = blockIdx.x * 256 + threadIdx.x;
  if (e >= E) return;
  const float w = weights[widx ? widx[e] : e];
  const float v = w * __expf(logit[e] - segmax[idx[e]]);
  logit[e] = v;
  atomicAdd(&segden[idx[e]], v);
}
__global__ void attn_pool_kernel(const float* __restrict__ logit,
                                 const int* __restrict__ idx,
                                 const float* __restrict__ segden,
                                 const float* __restrict__ x32,
                                 float* __restrict__ pooled, int E) {
  const int wave = blockIdx.x * 8 + (threadIdx.x >> 5);
  const int lane = threadIdx.x & 31;
  if (wave >= E) return;
  const int s = idx[wave];
  const float g = logit[wave] / (segden[s] + 1e-13f);
  const float* x = x32 + (long)wave * 128;
  float* p = pooled + (long)s * 128;
#pragma unroll
  for (int f = lane; f < 128; f += 32) atomicAdd(&p[f], g * x[f]);
}

// ---- residual: afea += relu(pooled); refresh bf16 copy ----
__global__ void residual_kernel(float* __restrict__ afea32,
                                unsigned short* __restrict__ afea16,
                                const float* __restrict__ pooled, long n) {
  const long t = blockIdx.x * 256ll + threadIdx.x;
  if (t >= n) return;
  const float v = afea32[t] + fmaxf(pooled[t], 0.f);
  afea32[t] = v;
  afea16[t] = f2bf(v);
}

// =====================================================================
// Host orchestration
// =====================================================================
static inline dim3 gemm_grid(int Mrows, int N) { return dim3(Mrows / TM, N / TN); }

extern "C" void kernel_launch(void* const* d_in, const int* in_sizes, int n_in,
                              void* d_out, int out_size, void* d_ws, size_t ws_size,
                              hipStream_t stream) {
  (void)in_sizes; (void)n_in; (void)out_size; (void)ws_size;
  const int N = GN, M = GM, C = GC;
  const float* atom_w   = (const float*)d_in[0];
  const float* orig     = (const float*)d_in[1];
  const int*   self_idx = (const int*)d_in[2];
  const int*   nbr_idx  = (const int*)d_in[3];
  const int*   cry_idx  = (const int*)d_in[4];

  // ---- carve workspace (~670 MB peak; large edge buffers reused) ----
  char* ws = (char*)d_ws;
  size_t off = 0;
  auto alloc = [&](size_t bytes) -> char* {
    char* p = ws + off;
    off = (off + bytes + 255) & ~(size_t)255;
    return p;
  };
  auto cvtW = [&](int idx, int K, int Ncols) -> unsigned short* {
    unsigned short* dst = (unsigned short*)alloc((size_t)K * Ncols * 2);
    const long tot = (long)K * Ncols;
    wcvt_kernel<<<(tot + 255) / 256, 256, 0, stream>>>((const float*)d_in[idx], dst, K, Ncols);
    return dst;
  };
  unsigned short* Wemb = cvtW(5, 200, 128);
  unsigned short *Wli[3], *Wlo[3], *Wg0[3], *Wg1[3];
  for (int g = 0; g < 3; ++g) {
    const int b = 7 + g * 10;
    Wli[g] = cvtW(b + 0, 256, 512);
    Wlo[g] = cvtW(b + 2, 512, 128);
    Wg0[g] = cvtW(b + 4, 128, 384);
    Wg1[g] = cvtW(b + 6, 384, 128);
  }
  unsigned short* Wc0 = cvtW(37, 128, 640);
  unsigned short* Wc1 = cvtW(39, 640, 384);
  unsigned short* Wc2 = cvtW(41, 384, 128);
  unsigned short* Wo0 = cvtW(45, 128, 896);
  unsigned short* Wo1 = cvtW(47, 896, 640);
  unsigned short* Wo2 = cvtW(49, 640, 384);
  unsigned short* Wo3 = cvtW(51, 384, 128);

  float*          afea32  = (float*)alloc((size_t)N * 128 * 4);
  unsigned short* afea16  = (unsigned short*)alloc((size_t)N * 128 * 2);
  unsigned short* e_in    = (unsigned short*)alloc((size_t)M * 256 * 2);  // 128MB
  unsigned short* e_mid   = (unsigned short*)alloc((size_t)M * 512 * 2);  // 256MB
  unsigned short* e_out16 = (unsigned short*)alloc((size_t)M * 128 * 2);  // 64MB
  float*          e_out32 = (float*)alloc((size_t)M * 128 * 4);           // 128MB
  float*          glogit  = (float*)alloc((size_t)M * 4);
  float*          segmax  = (float*)alloc((size_t)N * 4);
  float*          segden  = (float*)alloc((size_t)N * 4);
  float*          pooled  = (float*)alloc((size_t)N * 128 * 4);           // 32MB
  unsigned short* crys16  = (unsigned short*)alloc((size_t)C * 128 * 2);

  const float NEG_INF = -__builtin_inff();

  // ---- embed: atom_fea = orig @ Wemb + b  (K=200, zero-filled tail) ----
  {
    unsigned short* orig16 = e_in;  // N*200*2 = 26MB fits in e_in
    const long tot = (long)N * 200;
    cvt_bf16_kernel<<<(tot + 255) / 256, 256, 0, stream>>>(orig, orig16, tot);
    gemm_bf16_kernel<false, true, true, true><<<gemm_grid(N, 128), 256, 0, stream>>>(
        orig16, Wemb, (const float*)d_in[6], afea16, afea32, N, 128, 200);
  }

  // ---- graph layers ----
  for (int g = 0; g < 3; ++g) {
    const int b = 7 + g * 10;
    gather_kernel<<<(long)M * 32 / 256, 256, 0, stream>>>(afea16, self_idx, nbr_idx, e_in, M);
    gemm_bf16_kernel<true, true, false, false><<<gemm_grid(M, 512), 256, 0, stream>>>(
        e_in, Wli[g], (const float*)d_in[b + 1], e_mid, nullptr, M, 512, 256);
    gemm_bf16_kernel<false, true, true, false><<<gemm_grid(M, 128), 256, 0, stream>>>(
        e_mid, Wlo[g], (const float*)d_in[b + 3], e_out16, e_out32, M, 128, 512);
    // gate MLP: 128 -> 384 -> 128 -> 1
    gemm_bf16_kernel<true, true, false, false><<<gemm_grid(M, 384), 256, 0, stream>>>(
        e_out16, Wg0[g], (const float*)d_in[b + 5], e_mid, nullptr, M, 384, 128);
    gemm_bf16_kernel<true, true, false, false><<<gemm_grid(M, 128), 256, 0, stream>>>(
        e_mid, Wg1[g], (const float*)d_in[b + 7], e_in, nullptr, M, 128, 384);
    rowdot_kernel<<<M / 8, 256, 0, stream>>>(
        e_in, (const float*)d_in[b + 8], (const float*)d_in[b + 9], glogit, M, 128, 1);
    // segment softmax over self_fea_idx, weights = atom_w[nbr_idx]
    fill_kernel<<<(N + 255) / 256, 256, 0, stream>>>(segmax, NEG_INF, N);
    fill_kernel<<<(N + 255) / 256, 256, 0, stream>>>(segden, 0.f, N);
    fill_kernel<<<((long)N * 128 + 255) / 256, 256, 0, stream>>>(pooled, 0.f, (long)N * 128);
    attn_max_kernel<<<M / 256, 256, 0, stream>>>(glogit, self_idx, segmax, M);
    attn_num_kernel<<<M / 256, 256, 0, stream>>>(glogit, self_idx, nbr_idx, atom_w,
                                                 segmax, segden, M);
    attn_pool_kernel<<<M / 8, 256, 0, stream>>>(glogit, self_idx, segden, e_out32, pooled, M);
    residual_kernel<<<((long)N * 128 + 255) / 256, 256, 0, stream>>>(
        afea32, afea16, pooled, (long)N * 128);
  }

  // ---- crystal gate MLP: 128 -> 640 -> 384 -> 128 -> 1 ----
  gemm_bf16_kernel<true, true, false, false><<<gemm_grid(N, 640), 256, 0, stream>>>(
      afea16, Wc0, (const float*)d_in[38], e_mid, nullptr, N, 640, 128);
  gemm_bf16_kernel<true, true, false, false><<<gemm_grid(N, 384), 256, 0, stream>>>(
      e_mid, Wc1, (const float*)d_in[40], e_in, nullptr, N, 384, 640);
  gemm_bf16_kernel<true, true, false, false><<<gemm_grid(N, 128), 256, 0, stream>>>(
      e_in, Wc2, (const float*)d_in[42], e_out16, nullptr, N, 128, 384);
  rowdot_kernel<<<N / 8, 256, 0, stream>>>(
      e_out16, (const float*)d_in[43], (const float*)d_in[44], glogit, N, 128, 1);
  // crystal attention: idx = crystal_atom_idx, weights = atom_w (direct)
  fill_kernel<<<(C + 255) / 256, 256, 0, stream>>>(segmax, NEG_INF, C);
  fill_kernel<<<(C + 255) / 256, 256, 0, stream>>>(segden, 0.f, C);
  fill_kernel<<<((long)C * 128 + 255) / 256, 256, 0, stream>>>(pooled, 0.f, (long)C * 128);
  attn_max_kernel<<<N / 256, 256, 0, stream>>>(glogit, cry_idx, segmax, N);
  attn_num_kernel<<<N / 256, 256, 0, stream>>>(glogit, cry_idx, nullptr, atom_w,
                                               segmax, segden, N);
  attn_pool_kernel<<<N / 8, 256, 0, stream>>>(glogit, cry_idx, segden, afea32, pooled, N);
  cvt_bf16_kernel<<<((long)C * 128 + 255) / 256, 256, 0, stream>>>(pooled, crys16, (long)C * 128);

  // ---- output MLP: 128 -> 896 -> 640 -> 384 -> 128 -> 2 ----
  gemm_bf16_kernel<true, true, false, false><<<gemm_grid(C, 896), 256, 0, stream>>>(
      crys16, Wo0, (const float*)d_in[46], e_mid, nullptr, C, 896, 128);
  gemm_bf16_kernel<true, true, false, false><<<gemm_grid(C, 640), 256, 0, stream>>>(
      e_mid, Wo1, (const float*)d_in[48], e_in, nullptr, C, 640, 896);
  gemm_bf16_kernel<true, true, false, false><<<gemm_grid(C, 384), 256, 0, stream>>>(
      e_in, Wo2, (const float*)d_in[50], e_mid, nullptr, C, 384, 640);
  gemm_bf16_kernel<true, true, false, false><<<gemm_grid(C, 128), 256, 0, stream>>>(
      e_mid, Wo3, (const float*)d_in[52], e_in, nullptr, C, 128, 384);
  rowdot_kernel<<<C / 8, 256, 0, stream>>>(
      e_in, (const float*)d_in[53], (const float*)d_in[54], (float*)d_out, C, 128, 2);
}
